// NodeUpdateBlock_50861002719955
// MI455X (gfx1250) — matrix operand order repaired and checked by reading, not compiled, not run on hardware
//
#include <hip/hip_runtime.h>

// CDNA5 / gfx1250: wave32, v_wmma_f32_16x16x32_f16, TDM tensor_load_to_lds.
typedef __attribute__((ext_vector_type(16))) _Float16 v16h;
typedef __attribute__((ext_vector_type(8)))  float    v8f;
typedef __attribute__((ext_vector_type(4)))  unsigned v4u;
typedef __attribute__((ext_vector_type(4)))  int      v4i;
typedef __attribute__((ext_vector_type(8)))  int      v8i;

#define NN        100000
#define DIM       512
#define MUL       128
#define NA        10
#define LIN_SCALE 0.088388347648318447f   // 1/sqrt(128)
#define TP_SCALE  0.027950849718747371f   // 1/sqrt(1280)

// d_ws layout (f16 elements), all tiles fragment-packed for direct DMA to LDS:
//   Wl0 @ 0, Wl1 @ 16384, Wt0 @ 32768 (10 x 16384, a-major), Wt1 @ 196608
#define WS_WL1   16384
#define WS_WT0   32768
#define WS_WT1   196608
#define WS_ELEMS 360448

#if defined(__HIP_DEVICE_COMPILE__) && \
    __has_builtin(__builtin_amdgcn_tensor_load_to_lds) && \
    __has_builtin(__builtin_amdgcn_s_wait_tensorcnt)
#define TDM_OK 1
#else
#define TDM_OK 0
#endif

// fragment-packed index for a 128x128 f16 weight tile:
// B fragment (k-step kt, col, half-wave g) = ONE aligned v16h at (kt*128+col)*2+g
__device__ __forceinline__ int packed_idx(int k, int col) {
    return ((((k >> 5) << 7) + col) * 2 + ((k >> 4) & 1)) * 16 + (k & 15);
}

#if TDM_OK
__device__ __forceinline__ unsigned lds_byte_off(const void* p) {
    // LDS aperture: flat addr low 32 bits = byte offset in workgroup LDS
    return (unsigned)(unsigned long long)(size_t)p;
}

// 1-D TDM copy: nElems f16 elements, global -> LDS (ISA ch.8 D# bitfields).
__device__ __forceinline__ void tdm_load_lds(unsigned ldsOff, const _Float16* src, int nElems) {
    const unsigned long long ga = (unsigned long long)(size_t)src;
    v4u g0;
    g0.x = 1u;                                            // count=1, user mode
    g0.y = ldsOff;                                        // lds_addr
    g0.z = (unsigned)ga;                                  // global_addr[31:0]
    g0.w = (unsigned)((ga >> 32) & 0x01FFFFFFu) | (2u << 30);  // ga[56:32] | type=2
    v8i g1;
    g1[0] = (int)(1u << 16);                              // wg_mask=0, data_size=1 (2B)
    g1[1] = (int)(((unsigned)nElems & 0xFFFFu) << 16);    // tensor_dim0[15:0]
    g1[2] = (int)((((unsigned)nElems >> 16) & 0xFFFFu) | (1u << 16)); // dim0 hi | tensor_dim1=1
    g1[3] = (int)(((unsigned)nElems & 0xFFFFu) << 16);    // tile_dim0 = nElems
    g1[4] = 0;                                            // tile_dim1/2 = 0 (1-D tile)
    g1[5] = nElems;                                       // tensor_dim0_stride[31:0]
    g1[6] = 0;
    g1[7] = 0;
    v4i z4 = {0, 0, 0, 0};
#if __clang_major__ >= 23
    v8i z8 = {0, 0, 0, 0, 0, 0, 0, 0};
    __builtin_amdgcn_tensor_load_to_lds(g0, g1, z4, z4, z8, 0);
#else
    __builtin_amdgcn_tensor_load_to_lds(g0, g1, z4, z4, 0);   // ROCm 7.2 / clang-22
#endif
}
#endif // TDM_OK

// ---- one-time weight prepack: f32 -> f16, scales folded, fragment-packed ----
__global__ __launch_bounds__(256) void prepack_weights(
    const float* __restrict__ Wl0, const float* __restrict__ Wl1,
    const float* __restrict__ Wt0, const float* __restrict__ Wt1,
    _Float16* __restrict__ ws)
{
    const int i = (int)blockIdx.x * 256 + (int)threadIdx.x;
    if (i >= WS_ELEMS) return;
    float v; int dst;
    if (i < WS_WT0) {                                     // Wl0 / Wl1
        const float* W  = (i < WS_WL1) ? Wl0 : Wl1;
        const int base  = i & ~(MUL * MUL - 1);
        const int j     = i & (MUL * MUL - 1);
        const int k = j >> 7, col = j & 127;
        v   = W[k * MUL + col] * LIN_SCALE;
        dst = base + packed_idx(k, col);
    } else {                                              // Wt0 / Wt1
        const int t = i - WS_WT0;
        const bool second = t >= NA * MUL * MUL;
        const float* W = second ? Wt1 : Wt0;
        const int u = second ? t - NA * MUL * MUL : t;
        const int a = u >> 14, j = u & 16383;
        const int k = j >> 7, col = j & 127;
        v   = W[(size_t)k * (NA * MUL) + a * MUL + col] * TP_SCALE;
        dst = WS_WT0 + (second ? NA * MUL * MUL : 0) + (a << 14) + packed_idx(k, col);
    }
    ws[dst] = (_Float16)v;
}

// out_c = A_c @ (LIN_SCALE*Wl) + sum_a att[:,a] * (F_c @ TP_SCALE*Wt[:,a,:])
// Block: 128 nodes, 8 wave32s; wave = 16-node M-tile x all 128 out cols.
// USE_TDM: weights DMA'd from prepacked ws by wave 0, double-buffered behind WMMA.
template<bool IS_S, bool USE_TDM>
__global__ __launch_bounds__(256) void node_update_wmma(
    const float* __restrict__ m_i,
    const float* __restrict__ node_feats,
    const float* __restrict__ node_attrs,
    const float* __restrict__ Wl,          // (128,128)    [i][o]  (fallback path)
    const float* __restrict__ Wt,          // (128,10,128) [i][a][o]
    const _Float16* __restrict__ wsh,      // prepacked weights (TDM path)
    float* __restrict__ out)
{
    __shared__ v16h sWl[1024];                     // 32 KB
    __shared__ v16h sWt[USE_TDM ? 2048 : 1024];    // 64 KB double-buffer / 32 KB

    const int tid  = threadIdx.x;
    const int lane = tid & 31;
    const int wave = tid >> 5;
    const int l15  = lane & 15;
    const int g    = lane >> 4;
    const int xoff = IS_S ? 0 : (int)blockIdx.y;

    const _Float16* wsWl = wsh + (IS_S ? 0 : WS_WL1);
    const _Float16* wsWt = wsh + (IS_S ? WS_WT0 : WS_WT1);

    // ---- stage Wl (and kick off Wt[a=0] on the TDM path) ----
    if constexpr (USE_TDM) {
#if TDM_OK
        if (wave == 0) {
            tdm_load_lds(lds_byte_off(sWl), wsWl, MUL * MUL);
            tdm_load_lds(lds_byte_off(sWt), wsWt, MUL * MUL);   // buf0 <- Wt[0]
            __builtin_amdgcn_s_wait_tensorcnt(1);               // in-order: Wl done
        }
#else
        for (int idx = tid; idx < MUL * MUL / 2; idx += 256)
            ((unsigned*)sWl)[idx] = ((const unsigned*)wsWl)[idx];
#endif
        __syncthreads();
    } else {
        _Float16* sWlh = (_Float16*)sWl;
        for (int idx = tid; idx < MUL * MUL; idx += 256) {
            const int k = idx >> 7, col = idx & 127;
            sWlh[packed_idx(k, col)] = (_Float16)(Wl[k * MUL + col] * LIN_SCALE);
        }
        __syncthreads();
    }

    // ---- per-lane A/F row (A layout: row = lane&15, kbase = g ? 16 : 0) ----
    const int noderaw = (int)blockIdx.x * 128 + wave * 16 + l15;
    const int node    = noderaw < NN ? noderaw : NN - 1;     // clamp, never OOB
    const float* arow = m_i        + (size_t)node * DIM + (IS_S ? 0 : MUL + xoff);
    const float* frow = node_feats + (size_t)node * DIM + (IS_S ? 0 : MUL + xoff);
    const int kbaseA  = g ? 16 : 0;
    __builtin_prefetch(frow, 0, 3);                          // global_prefetch_b8

    v8f acc[8] = {};

    // ================= linear part: A(m_i) @ Wl =================
    #pragma unroll
    for (int kt = 0; kt < 4; ++kt) {
        v16h afrag;
        #pragma unroll
        for (int e = 0; e < 16; ++e) {
            const int k = kt * 32 + kbaseA + e;
            afrag[e] = (_Float16)arow[IS_S ? k : 3 * k];
        }
        #pragma unroll
        for (int ct = 0; ct < 8; ++ct) {
            const v16h bfrag = sWl[((kt << 7) + (ct << 4) + l15) * 2 + g];
            acc[ct] = __builtin_amdgcn_wmma_f32_16x16x32_f16(
                false, afrag, false, bfrag, (short)0, acc[ct], false, false);
        }
    }

    // ---- preload F fragments once; reused for all 10 attention terms ----
    v16h ffrag[4];
    #pragma unroll
    for (int kt = 0; kt < 4; ++kt) {
        #pragma unroll
        for (int e = 0; e < 16; ++e) {
            const int k = kt * 32 + kbaseA + e;
            ffrag[kt][e] = (_Float16)frow[IS_S ? k : 3 * k];
        }
    }

    // ================= tensor-product part =================
    for (int a = 0; a < NA; ++a) {
        const v16h* buf;
        if constexpr (USE_TDM) {
#if TDM_OK
            if (wave == 0) __builtin_amdgcn_s_wait_tensorcnt(0);   // buf[a&1] filled
            __syncthreads();       // publish buf[a&1]; all reads of buf[(a+1)&1] done
            if ((wave == 0) && (a + 1 < NA))                       // prefetch next tile,
                tdm_load_lds(lds_byte_off(sWt + ((a + 1) & 1) * 1024),
                             wsWt + (a + 1) * (MUL * MUL), MUL * MUL); // overlaps WMMAs
            buf = sWt + (a & 1) * 1024;
#else
            __syncthreads();
            for (int idx = tid; idx < MUL * MUL / 2; idx += 256)
                ((unsigned*)sWt)[idx] = ((const unsigned*)(wsWt + a * (MUL * MUL)))[idx];
            __syncthreads();
            buf = sWt;
#endif
        } else {
            __syncthreads();
            _Float16* sWth = (_Float16*)sWt;
            for (int idx = tid; idx < MUL * MUL; idx += 256) {
                const int k = idx >> 7, col = idx & 127;
                sWth[packed_idx(k, col)] =
                    (_Float16)(Wt[(size_t)k * (NA * MUL) + a * MUL + col] * TP_SCALE);
            }
            __syncthreads();
            buf = sWt;
        }

        const _Float16 av = (_Float16)node_attrs[(size_t)node * NA + a]; // one-hot: exact
        #pragma unroll
        for (int kt = 0; kt < 4; ++kt) {
            v16h afrag;
            #pragma unroll
            for (int e = 0; e < 16; ++e) afrag[e] = ffrag[kt][e] * av;   // v_pk_mul_f16
            #pragma unroll
            for (int ct = 0; ct < 8; ++ct) {
                const v16h bfrag = buf[((kt << 7) + (ct << 4) + l15) * 2 + g];
                acc[ct] = __builtin_amdgcn_wmma_f32_16x16x32_f16(
                    false, afrag, false, bfrag, (short)0, acc[ct], false, false);
            }
        }
    }

    // ===== store (C/D layout: VGPR r -> row rbase+r, col = ct*16 + lane&15) =====
    const int rbase = g ? 8 : 0;
    #pragma unroll
    for (int r = 0; r < 8; ++r) {
        const int rowNode = (int)blockIdx.x * 128 + wave * 16 + rbase + r;
        if (rowNode < NN) {
            float* orow = out + (size_t)rowNode * DIM + (IS_S ? 0 : MUL + xoff);
            #pragma unroll
            for (int ct = 0; ct < 8; ++ct) {
                const int col = ct * 16 + l15;
                orow[IS_S ? col : 3 * col] = acc[ct][r];
            }
        }
    }
}

extern "C" void kernel_launch(void* const* d_in, const int* in_sizes, int n_in,
                              void* d_out, int out_size, void* d_ws, size_t ws_size,
                              hipStream_t stream) {
    const float* m_i = (const float*)d_in[0];
    const float* nf  = (const float*)d_in[1];
    const float* na  = (const float*)d_in[2];
    const float* Wl0 = (const float*)d_in[3];
    const float* Wl1 = (const float*)d_in[4];
    const float* Wt0 = (const float*)d_in[5];
    const float* Wt1 = (const float*)d_in[6];
    float* out = (float*)d_out;

    const int nblk = (NN + 127) / 128;
    const bool use_tdm = ws_size >= (size_t)WS_ELEMS * 2u;   // 704 KB of f16 scratch

    if (use_tdm) {
        _Float16* ws = (_Float16*)d_ws;
        prepack_weights<<<WS_ELEMS / 256, 256, 0, stream>>>(Wl0, Wl1, Wt0, Wt1, ws);
        node_update_wmma<true , true ><<<dim3(nblk, 1), 256, 0, stream>>>(m_i, nf, na, Wl0, Wt0, ws, out);
        node_update_wmma<false, true ><<<dim3(nblk, 3), 256, 0, stream>>>(m_i, nf, na, Wl1, Wt1, ws, out);
    } else {
        node_update_wmma<true , false><<<dim3(nblk, 1), 256, 0, stream>>>(m_i, nf, na, Wl0, Wt0, nullptr, out);
        node_update_wmma<false, false><<<dim3(nblk, 3), 256, 0, stream>>>(m_i, nf, na, Wl1, Wt1, nullptr, out);
    }
    (void)in_sizes; (void)n_in; (void)out_size;
}
